// PointNetConv_17197049053636
// MI455X (gfx1250) — matrix operand order repaired and checked by reading, not compiled, not run on hardware
//
#include <hip/hip_runtime.h>
#include <hip/hip_bf16.h>

// ---------------------------------------------------------------------------
// PointNet++ set-abstraction: FPS -> ball query -> group -> 3xMLP(WMMA) -> max
// B=16, N=4096, NPOINT=1024, NSAMPLE=32, RADIUS=0.2, C_FEAT=64, MLP 64/64/128
// ---------------------------------------------------------------------------

#define BQ   16
#define NQ   4096
#define NPQ  1024
#define NSQ  32
#define CFT  64
#define CIN0 96   // 3+64=67 padded to 96 (3 K-chunks of 32)

typedef __attribute__((ext_vector_type(16))) _Float16 v16h;
typedef __attribute__((ext_vector_type(8)))  _Float16 v8h;
typedef __attribute__((ext_vector_type(8)))  float    v8f;

union V16U { v16h v; v8h h[2]; };

// A-matrix fragment (16-bit, 16x32): lane half selects K base {0,8}; per lane
// halves [0..7] = row[k0..k0+7], halves [8..15] = row[k0+16..k0+23].
__device__ __forceinline__ v16h ld_frag_a(const _Float16* row, int k0) {
    V16U u;
    u.h[0] = *(const v8h*)(row + k0);
    u.h[1] = *(const v8h*)(row + k0 + 16);
    return u.v;
}

// B-matrix fragment (32x16): lane half selects K base {0,8}; halves [0..7] =
// W^T row[k0..k0+7], halves [8..15] = row[k0+16..k0+23] (column of B).
__device__ __forceinline__ v16h ld_frag_b(const _Float16* row, int k0) {
    V16U u;
    u.h[0] = *(const v8h*)(row + k0);
    u.h[1] = *(const v8h*)(row + k0 + 16);
    return u.v;
}

// One MLP layer for a 32-row activation tile: Y[32xCOUT] = relu(X[32xCIN] @ W)
// Wt is W^T ([COUT][CIN], f16). X/Y live in LDS, Wt in global (L2-resident).
template<int CIN, int COUT>
__device__ __forceinline__ void mlp_layer(const _Float16* __restrict__ X, int ldx,
                                          const _Float16* __restrict__ Wt,
                                          _Float16* __restrict__ Y, int ldy,
                                          int lane) {
    const int lid  = lane & 15;
    const int half = (lane >> 4) & 1;
#pragma unroll
    for (int t = 0; t < 2; ++t) {                       // M tiles (rows 16t..)
        const _Float16* arow = X + (t * 16 + lid) * ldx;
#pragma unroll
        for (int j = 0; j < COUT / 16; ++j) {           // N tiles
            const _Float16* brow = Wt + (j * 16 + lid) * CIN;
            v8f acc = {};
#pragma unroll
            for (int kc = 0; kc < CIN / 32; ++kc) {     // K chunks of 32
                v16h a = ld_frag_a(arow, kc * 32 + half * 8);
                v16h b = ld_frag_b(brow, kc * 32 + half * 8);
                acc = __builtin_amdgcn_wmma_f32_16x16x32_f16(
                    false, a, false, b, (short)0, acc, false, false);
            }
            // C/D layout: VGPR r -> M = r + 8*half, N = lid
#pragma unroll
            for (int r = 0; r < 8; ++r) {
                float v = acc[r];
                v = v > 0.f ? v : 0.f;                  // ReLU
                int m = t * 16 + half * 8 + r;
                Y[m * ldy + j * 16 + lid] = (_Float16)v;
            }
        }
    }
}

// ---------------------------------------------------------------------------
// Kernel 0: weights -> f16 transposed (+K padding for layer 0)
// w0t [64][96], w1t [64][64], w2t [128][64]
// ---------------------------------------------------------------------------
__global__ __launch_bounds__(256) void wconv_kernel(
        const float* __restrict__ w0, const float* __restrict__ w1,
        const float* __restrict__ w2,
        _Float16* __restrict__ w0t, _Float16* __restrict__ w1t,
        _Float16* __restrict__ w2t) {
    int t = blockIdx.x * 256 + threadIdx.x;
    if (t < 64 * CIN0) {
        int o = t / CIN0, i = t % CIN0;
        w0t[t] = (i < 67) ? (_Float16)w0[i * 64 + o] : (_Float16)0.f;
    }
    if (t < 64 * 64) {
        int o = t / 64, i = t % 64;
        w1t[t] = (_Float16)w1[i * 64 + o];
    }
    if (t < 128 * 64) {
        int o = t / 64, i = t % 64;
        w2t[t] = (_Float16)w2[i * 128 + o];
    }
}

// ---------------------------------------------------------------------------
// Kernel 1: farthest point sampling, one block per batch.
// Writes fps new_xyz into d_out[0:B*NP*3] and workspace copy.
// ---------------------------------------------------------------------------
__global__ __launch_bounds__(256) void fps_kernel(
        const float* __restrict__ xyz,
        float* __restrict__ newxyz_out, float* __restrict__ newxyz_ws) {
    __shared__ float dists[NQ];
    __shared__ float red_v[256];
    __shared__ int   red_i[256];
    __shared__ float lastp[3];

    const int b   = blockIdx.x;
    const int tid = threadIdx.x;
    const float* x = xyz + (size_t)b * NQ * 3;

    for (int i = tid; i < NQ; i += 256) dists[i] = 1e10f;

    int last = 0;
    if (tid == 0) {
        float px = x[0], py = x[1], pz = x[2];
        size_t o = (size_t)b * NPQ * 3;
        newxyz_out[o + 0] = px; newxyz_out[o + 1] = py; newxyz_out[o + 2] = pz;
        newxyz_ws [o + 0] = px; newxyz_ws [o + 1] = py; newxyz_ws [o + 2] = pz;
    }

    for (int s = 1; s < NPQ; ++s) {
        if (tid == 0) {
            lastp[0] = x[last * 3 + 0];
            lastp[1] = x[last * 3 + 1];
            lastp[2] = x[last * 3 + 2];
        }
        __syncthreads();
        float lx = lastp[0], ly = lastp[1], lz = lastp[2];
        float best = -1.f; int bi = 0x7fffffff;
        for (int i = tid; i < NQ; i += 256) {
            float dx = x[i * 3 + 0] - lx;
            float dy = x[i * 3 + 1] - ly;
            float dz = x[i * 3 + 2] - lz;
            float d  = dx * dx + dy * dy + dz * dz;
            float dd = fminf(dists[i], d);
            dists[i] = dd;
            if (dd > best) { best = dd; bi = i; }
        }
        red_v[tid] = best; red_i[tid] = bi;
        __syncthreads();
        for (int o = 128; o > 0; o >>= 1) {
            if (tid < o) {
                float ov = red_v[tid + o]; int oi = red_i[tid + o];
                if (ov > red_v[tid] || (ov == red_v[tid] && oi < red_i[tid])) {
                    red_v[tid] = ov; red_i[tid] = oi;
                }
            }
            __syncthreads();
        }
        last = red_i[0];
        if (tid == 0) {
            float px = x[last * 3 + 0], py = x[last * 3 + 1], pz = x[last * 3 + 2];
            size_t o = ((size_t)b * NPQ + s) * 3;
            newxyz_out[o + 0] = px; newxyz_out[o + 1] = py; newxyz_out[o + 2] = pz;
            newxyz_ws [o + 0] = px; newxyz_ws [o + 1] = py; newxyz_ws [o + 2] = pz;
        }
        __syncthreads();
    }
}

// ---------------------------------------------------------------------------
// Kernel 2: ball query. One thread per query point; xyz[b] staged in LDS.
// First NSAMPLE in-ball in original order, padded with first hit (0 if none).
// ---------------------------------------------------------------------------
__global__ __launch_bounds__(256) void ballq_kernel(
        const float* __restrict__ xyz, const float* __restrict__ newxyz_ws,
        int* __restrict__ idx_ws) {
    __shared__ float sx[NQ * 3];
    const int b = blockIdx.y;
    const float* x = xyz + (size_t)b * NQ * 3;
    for (int i = threadIdx.x; i < NQ * 3; i += 256) sx[i] = x[i];
    __syncthreads();

    const int p = blockIdx.x * 256 + threadIdx.x;
    const int q = b * NPQ + p;
    const float cx = newxyz_ws[q * 3 + 0];
    const float cy = newxyz_ws[q * 3 + 1];
    const float cz = newxyz_ws[q * 3 + 2];
    const int base = q * NSQ;

    int cnt = 0, first = 0;
    for (int j = 0; j < NQ; ++j) {
        float dx = sx[j * 3 + 0] - cx;
        float dy = sx[j * 3 + 1] - cy;
        float dz = sx[j * 3 + 2] - cz;
        if (dx * dx + dy * dy + dz * dz < 0.04f) {
            if (cnt == 0) first = j;
            idx_ws[base + cnt] = j;
            if (++cnt == NSQ) break;
        }
    }
    for (int k = cnt; k < NSQ; ++k) idx_ws[base + k] = first;
}

// ---------------------------------------------------------------------------
// Kernel 3: gather + 3-layer WMMA MLP + max-pool. One query point per wave32,
// 4 waves per block. Activations in LDS; weight fragments straight from L2.
// ---------------------------------------------------------------------------
#define WAVES 4
__global__ __launch_bounds__(WAVES * 32) void group_mlp_kernel(
        const float* __restrict__ xyz, const float* __restrict__ points,
        const float* __restrict__ newxyz_ws, const int* __restrict__ idx_ws,
        const _Float16* __restrict__ w0t, const _Float16* __restrict__ w1t,
        const _Float16* __restrict__ w2t, float* __restrict__ out_points) {
    __shared__ __align__(32) _Float16 sA[WAVES][NSQ * CIN0];   // feat / h1
    __shared__ __align__(32) _Float16 sB[WAVES][NSQ * 128];    // h0 / h2

    const int wave = threadIdx.x >> 5;
    const int lane = threadIdx.x & 31;
    const int q    = blockIdx.x * WAVES + wave;   // 0 .. B*NPOINT-1
    const int b    = q >> 10;                     // / NPOINT
    _Float16* A  = sA[wave];
    _Float16* Bm = sB[wave];

    // ---- build feat rows: lane s owns sample s ----------------------------
    const int   i  = idx_ws[q * NSQ + lane];
    const float cx = newxyz_ws[q * 3 + 0];
    const float cy = newxyz_ws[q * 3 + 1];
    const float cz = newxyz_ws[q * 3 + 2];
    const float* pt = xyz + ((size_t)b * NQ + i) * 3;
    _Float16* arow = A + lane * CIN0;
    arow[0] = (_Float16)(pt[0] - cx);
    arow[1] = (_Float16)(pt[1] - cy);
    arow[2] = (_Float16)(pt[2] - cz);
    const float* f = points + ((size_t)b * NQ + i) * CFT;
#pragma unroll
    for (int c = 0; c < CFT; ++c) arow[3 + c] = (_Float16)f[c];
#pragma unroll
    for (int c = 67; c < CIN0; ++c) arow[c] = (_Float16)0.f;
    __syncthreads();

    // ---- MLP: 96->64 ; 64->64 ; 64->128 (ReLU each, f32 accum) ------------
    mlp_layer<CIN0, 64>(A, CIN0, w0t, Bm, 64, lane);
    __syncthreads();
    mlp_layer<64, 64>(Bm, 64, w1t, A, CIN0, lane);   // h1 into A (ld 96)
    __syncthreads();
    mlp_layer<64, 128>(A, CIN0, w2t, Bm, 128, lane);
    __syncthreads();

    // ---- max over NSAMPLE rows; 4 output channels per lane ----------------
    float* outp = out_points + (size_t)q * 128;
#pragma unroll
    for (int cc = 0; cc < 4; ++cc) {
        int c = lane * 4 + cc;
        float mx = (float)Bm[c];
        for (int m = 1; m < NSQ; ++m) mx = fmaxf(mx, (float)Bm[m * 128 + c]);
        outp[c] = mx;
    }
}

// ---------------------------------------------------------------------------
extern "C" void kernel_launch(void* const* d_in, const int* in_sizes, int n_in,
                              void* d_out, int out_size, void* d_ws, size_t ws_size,
                              hipStream_t stream) {
    const float* xyz    = (const float*)d_in[0];   // [16,4096,3]
    const float* points = (const float*)d_in[1];   // [16,4096,64]
    const float* w0     = (const float*)d_in[2];   // [67,64]
    const float* w1     = (const float*)d_in[3];   // [64,64]
    const float* w2     = (const float*)d_in[4];   // [64,128]

    float* newxyz_out = (float*)d_out;                         // [16,1024,3]
    float* newpts_out = (float*)d_out + (size_t)BQ * NPQ * 3;  // [16,1024,128]

    char* ws = (char*)d_ws;
    int*       fps_idx   = (int*)(ws + 0);                 // unused after fps (kept for clarity)
    float*     newxyz_ws = (float*)(ws + 65536);           // 16*1024*3 f32
    int*       idx_ws    = (int*)(ws + 262144);            // 16*1024*32 i32
    _Float16*  w0t       = (_Float16*)(ws + 2359296);      // [64][96]
    _Float16*  w1t       = (_Float16*)(ws + 2371584);      // [64][64]
    _Float16*  w2t       = (_Float16*)(ws + 2379776);      // [128][64]
    (void)fps_idx; (void)in_sizes; (void)n_in; (void)out_size; (void)ws_size;

    wconv_kernel<<<32, 256, 0, stream>>>(w0, w1, w2, w0t, w1t, w2t);
    fps_kernel<<<BQ, 256, 0, stream>>>(xyz, newxyz_out, newxyz_ws);
    ballq_kernel<<<dim3(NPQ / 256, BQ), 256, 0, stream>>>(xyz, newxyz_ws, idx_ws);
    group_mlp_kernel<<<(BQ * NPQ) / WAVES, WAVES * 32, 0, stream>>>(
        xyz, points, newxyz_ws, idx_ws, w0t, w1t, w2t, newpts_out);
}
